// GNNEncoder_65085934404103
// MI455X (gfx1250) — compile-verified, matrix operand order checked by reading, not run on hardware
//
#include <hip/hip_runtime.h>
#include <cstdint>
#include <cstddef>

typedef __bf16 bf16;
typedef __attribute__((ext_vector_type(8)))  bf16  v8bf;
typedef __attribute__((ext_vector_type(16))) bf16  v16bf;
typedef __attribute__((ext_vector_type(8)))  float v8f;
typedef __attribute__((ext_vector_type(4)))  float v4f;

#define DEVINL __device__ __forceinline__

// ---- CDNA5 async global->LDS copy (ASYNCcnt-tracked, no VGPR data) ----
// ISA 15.18.3: GLOBAL_LOAD_ASYNC_TO_LDS_B128  vdst(=LDS byte addr), vaddr, off
DEVINL void async_copy_b128(const void* gsrc, void* ldst) {
  unsigned lds_off = (unsigned)(unsigned long long)ldst;  // generic->LDS offset (addr[31:0])
  asm volatile("global_load_async_to_lds_b128 %0, %1, off"
               :: "v"(lds_off), "v"(gsrc) : "memory");
}
DEVINL void wait_asynccnt0() {
  asm volatile("s_wait_asynccnt 0x0" ::: "memory");
}

// Load a 16-lane-row bf16 WMMA fragment (16x32 A or 32x16 B) from LDS.
// Per ISA 7.12.2: lane%16 = matrix row (A) / col (B); lane/16 = h selects
// K-halves: VGPR0-3 hold K = 8h..8h+7, VGPR4-7 hold K = 16+8h..16+8h+7.
DEVINL v16bf frag16(const bf16* row, int k0, int h) {
  v8bf lo = *(const v8bf*)(row + k0 + 8 * h);
  v8bf hi = *(const v8bf*)(row + k0 + 16 + 8 * h);
  return __builtin_shufflevector(lo, hi, 0, 1, 2, 3, 4, 5, 6, 7,
                                 8, 9, 10, 11, 12, 13, 14, 15);
}

// One-time weight prep: W[K][128] f32 -> Wt[128][K] bf16 (transposed, packed).
__global__ __launch_bounds__(256) void prep_w_kernel(
    const float* __restrict__ W, unsigned short* __restrict__ Wt, int K) {
  int idx = blockIdx.x * 256 + threadIdx.x;
  if (idx >= K * 128) return;
  int k = idx >> 7, n = idx & 127;
  Wt[(size_t)n * K + k] = __builtin_bit_cast(unsigned short, (bf16)W[idx]);
}

// C[128 x 128] = A[nrows x K](bf16) * W[K x 128] + bias, for NW pre-packed
// transposed weights sharing one async-staged A tile. Whole A tile and whole
// W^T live in LDS; inner loop is pure ds_load_b128 + v_wmma_f32_16x16x32_bf16.
template <int K, int NW, bool NT>
__global__ __launch_bounds__(256) void wmma_gemm_kernel(
    const unsigned short* __restrict__ A, int nrows,
    const unsigned short* __restrict__ Wt0, const unsigned short* __restrict__ Wt1,
    const unsigned short* __restrict__ Wt2, const unsigned short* __restrict__ Wt3,
    const float* __restrict__ b0, const float* __restrict__ b1,
    const float* __restrict__ b2, const float* __restrict__ b3,
    float* __restrict__ C0, float* __restrict__ C1,
    float* __restrict__ C2, float* __restrict__ C3) {
  __shared__ bf16 sA[128 * K];  // 128 rows x K bf16 (K=128 -> 32 KB)
  __shared__ bf16 sB[128 * K];  // W^T: 128 cols x K bf16

  const int tid = threadIdx.x;
  const int m0 = blockIdx.x * 128;
  const int CPR = K / 8;  // 16B chunks per row

  // Stage A tile: per-lane 16-byte async copies straight into LDS.
  for (int c = tid; c < 128 * CPR; c += 256) {
    int row = c / CPR;
    int col = (c % CPR) * 8;
    int g = m0 + row;
    if (g < nrows) {
      async_copy_b128(A + (size_t)g * K + col, &sA[row * K + col]);
    } else {
      *(uint4*)&sA[row * K + col] = make_uint4(0u, 0u, 0u, 0u);
    }
  }
  wait_asynccnt0();

  const unsigned short* Wl[4] = {Wt0, Wt1, Wt2, Wt3};
  const float* bl[4] = {b0, b1, b2, b3};
  float* Cl[4] = {C0, C1, C2, C3};

  const int wv = tid >> 5;
  const int lane = tid & 31;
  const int h = lane >> 4;
  const int nl = lane & 15;
  const bf16* arow = sA + (size_t)(wv * 16 + nl) * K;

  for (int wi = 0; wi < NW; ++wi) {
    __syncthreads();  // WAR on sB; first pass also covers A-staging visibility
    const unsigned short* Wt = Wl[wi];
    for (int c = tid; c < 128 * CPR; c += 256) {
      int n = c / CPR;
      int col = (c % CPR) * 8;
      async_copy_b128(Wt + (size_t)n * K + col, &sB[n * K + col]);
    }
    wait_asynccnt0();
    __syncthreads();

    v8f acc[8];
#pragma unroll
    for (int t = 0; t < 8; ++t)
#pragma unroll
      for (int i = 0; i < 8; ++i) acc[t][i] = 0.f;

#pragma unroll
    for (int ks = 0; ks < K / 32; ++ks) {
      const int k0 = ks * 32;
      v16bf af = frag16(arow, k0, h);
#pragma unroll
      for (int t = 0; t < 8; ++t) {
        v16bf bfr = frag16(sB + (size_t)(t * 16 + nl) * K, k0, h);
        acc[t] = __builtin_amdgcn_wmma_f32_16x16x32_bf16(
            false, af, false, bfr, (short)0, acc[t], false, false);
      }
    }

    // D layout (ISA 7.12.2): lane%16 = N, VGPR r -> M = r + 8*(lane/16).
    const float* bias = bl[wi];
    float* C = Cl[wi];
#pragma unroll
    for (int t = 0; t < 8; ++t) {
      float bn = bias ? bias[t * 16 + nl] : 0.f;
#pragma unroll
      for (int r = 0; r < 8; ++r) {
        int m = m0 + wv * 16 + r + 8 * h;
        if (m < nrows) {
          float val = acc[t][r] + bn;
          float* p = &C[(size_t)m * 128 + t * 16 + nl];
          if (NT)
            __builtin_nontemporal_store(val, p);  // streamed output (ek)
          else
            *p = val;
        }
      }
    }
  }
}

// f32 -> bf16 pack (4 elements/thread), optional fused ReLU.
__global__ __launch_bounds__(256) void cvt_bf16_kernel(
    const float* __restrict__ in, unsigned short* __restrict__ out,
    size_t n4, int relu) {
  size_t i = (size_t)blockIdx.x * 256 + threadIdx.x;
  if (i >= n4) return;
  float4 f = ((const float4*)in)[i];
  if (relu) {
    f.x = fmaxf(f.x, 0.f); f.y = fmaxf(f.y, 0.f);
    f.z = fmaxf(f.z, 0.f); f.w = fmaxf(f.w, 0.f);
  }
  ushort4 o;
  o.x = __builtin_bit_cast(unsigned short, (bf16)f.x);
  o.y = __builtin_bit_cast(unsigned short, (bf16)f.y);
  o.z = __builtin_bit_cast(unsigned short, (bf16)f.z);
  o.w = __builtin_bit_cast(unsigned short, (bf16)f.w);
  ((ushort4*)out)[i] = o;
}

DEVINL void atomicMaxF(float* addr, float val) {
  unsigned* u = (unsigned*)addr;
  unsigned cur = *u;
  while (__uint_as_float(cur) < val) {
    unsigned prev = atomicCAS(u, cur, __float_as_uint(val));
    if (prev == cur) break;
    cur = prev;
  }
}

__global__ __launch_bounds__(256) void init_nodes_kernel(float* __restrict__ amax,
                                                         float* __restrict__ denom,
                                                         int n) {
  int i = blockIdx.x * 256 + threadIdx.x;
  if (i < n) {
    amax[i] = -__builtin_inff();
    denom[i] = 0.f;
  }
}

// One wave32 per edge: alpha = q[dst].(k[src]+ek)/sqrt(d); atomic segment-max.
// ek is a >L2-sized stream -> NT loads so the q/k/v gather tables stay in L2.
__global__ __launch_bounds__(256) void edge_alpha_kernel(
    const int* __restrict__ src, const int* __restrict__ dst,
    const float* __restrict__ q, const float* __restrict__ k,
    const float* __restrict__ ek, float* __restrict__ alpha,
    float* __restrict__ amax, int E, float scale) {
  int wv = threadIdx.x >> 5, lane = threadIdx.x & 31;
  int e = blockIdx.x * 8 + wv;
  if (e >= E) return;
  if (e + 64 < E)  // stream-ahead on the big ek matrix -> global_prefetch_b8
    __builtin_prefetch(ek + (size_t)(e + 64) * 128 + lane * 4, 0, 1);
  int s = src[e], d = dst[e];
  float4 qv = ((const float4*)(q + (size_t)d * 128))[lane];
  float4 kv = ((const float4*)(k + (size_t)s * 128))[lane];
  v4f ev = __builtin_nontemporal_load((const v4f*)(ek + (size_t)e * 128) + lane);
  float sum = qv.x * (kv.x + ev.x) + qv.y * (kv.y + ev.y) +
              qv.z * (kv.z + ev.z) + qv.w * (kv.w + ev.w);
#pragma unroll
  for (int off = 16; off > 0; off >>= 1) sum += __shfl_down(sum, off, 32);
  if (lane == 0) {
    float al = sum * scale;
    alpha[e] = al;
    atomicMaxF(&amax[d], al);
  }
}

__global__ __launch_bounds__(256) void edge_softmax_kernel(
    const int* __restrict__ dst, float* __restrict__ alpha,
    const float* __restrict__ amax, float* __restrict__ denom, int E) {
  int e = blockIdx.x * 256 + threadIdx.x;
  if (e >= E) return;
  int d = dst[e];
  float ex = __expf(alpha[e] - amax[d]);
  alpha[e] = ex;  // reuse buffer: alpha now holds exp values
  atomicAdd(&denom[d], ex);
}

// One wave32 per edge: out[dst] += w * (v[src] + ek)   (f32 global atomics)
__global__ __launch_bounds__(256) void edge_agg_kernel(
    const int* __restrict__ src, const int* __restrict__ dst,
    const float* __restrict__ v, const float* __restrict__ ek,
    const float* __restrict__ ex, const float* __restrict__ denom,
    float* __restrict__ out, int E) {
  int wv = threadIdx.x >> 5, lane = threadIdx.x & 31;
  int e = blockIdx.x * 8 + wv;
  if (e >= E) return;
  if (e + 64 < E)
    __builtin_prefetch(ek + (size_t)(e + 64) * 128 + lane * 4, 0, 1);
  int s = src[e], d = dst[e];
  float w = ex[e] / fmaxf(denom[d], 1e-16f);
  float4 vv = ((const float4*)(v + (size_t)s * 128))[lane];
  v4f ev = __builtin_nontemporal_load((const v4f*)(ek + (size_t)e * 128) + lane);
  float* o = out + (size_t)d * 128 + lane * 4;
  atomicAdd(o + 0, w * (vv.x + ev.x));
  atomicAdd(o + 1, w * (vv.y + ev.y));
  atomicAdd(o + 2, w * (vv.z + ev.z));
  atomicAdd(o + 3, w * (vv.w + ev.w));
}

extern "C" void kernel_launch(void* const* d_in, const int* in_sizes, int n_in,
                              void* d_out, int out_size, void* d_ws, size_t ws_size,
                              hipStream_t stream) {
  const int N = 50000, E = 640000;
  const float* x = (const float*)d_in[0];
  const int* ei = (const int*)d_in[1];
  const float* ea = (const float*)d_in[2];
  const float *Wq1 = (const float*)d_in[3], *bq1 = (const float*)d_in[4];
  const float *Wk1 = (const float*)d_in[5], *bk1 = (const float*)d_in[6];
  const float *Wv1 = (const float*)d_in[7], *bv1 = (const float*)d_in[8];
  const float* We1 = (const float*)d_in[9];
  const float *Ws1 = (const float*)d_in[10], *bs1 = (const float*)d_in[11];
  const float *Wq2 = (const float*)d_in[12], *bq2 = (const float*)d_in[13];
  const float *Wk2 = (const float*)d_in[14], *bk2 = (const float*)d_in[15];
  const float *Wv2 = (const float*)d_in[16], *bv2 = (const float*)d_in[17];
  const float* We2 = (const float*)d_in[18];
  const float *Ws2 = (const float*)d_in[19], *bs2 = (const float*)d_in[20];
  const int* srcI = ei;
  const int* dstI = ei + E;

  float* ws = (float*)d_ws;
  size_t off = 0;
  float* ek = ws + off;   off += (size_t)E * 128;
  float* qb = ws + off;   off += (size_t)N * 128;
  float* kb = ws + off;   off += (size_t)N * 128;
  float* vb = ws + off;   off += (size_t)N * 128;
  float* hb = ws + off;   off += (size_t)N * 128;
  float* alph = ws + off; off += (size_t)E;
  float* amax = ws + off; off += (size_t)N;
  float* deno = ws + off; off += (size_t)N;
  // bf16 staging copies of the GEMM A-matrices
  unsigned short* xbf  = (unsigned short*)(ws + off);
  unsigned short* eabf = xbf + (size_t)N * 128;
  unsigned short* hbf  = eabf + (size_t)E * 64;
  // pre-packed transposed bf16 weights [128][K]
  unsigned short* wt = hbf + (size_t)N * 128;
  unsigned short* wtN[8];
  for (int i = 0; i < 8; ++i) wtN[i] = wt + (size_t)i * 128 * 128;
  unsigned short* wtE1 = wt + (size_t)8 * 128 * 128;
  unsigned short* wtE2 = wtE1 + (size_t)128 * 64;
  float* outp = (float*)d_out;

  const float scale = 0.08838834764831845f;  // 1/sqrt(128)
  const int gemmN = (N + 127) / 128;
  const int gemmE = (E + 127) / 128;
  const int blkE8 = (E + 7) / 8;
  const int blkE = (E + 255) / 256;
  const int blkN = (N + 255) / 256;
  const size_t xq = (size_t)N * 128 / 4;  // float4 groups in node matrices
  const size_t eq = (size_t)E * 64 / 4;
  const int blkXq = (int)((xq + 255) / 256);
  const int blkEq = (int)((eq + 255) / 256);
  const int blkW128 = (128 * 128 + 255) / 256;
  const int blkW64 = (64 * 128 + 255) / 256;

  // One-time preps: bf16 packs of A-inputs + transposed bf16 weights.
  cvt_bf16_kernel<<<blkXq, 256, 0, stream>>>(x, xbf, xq, 0);
  cvt_bf16_kernel<<<blkEq, 256, 0, stream>>>(ea, eabf, eq, 0);
  const float* Wsrc[8] = {Wq1, Wk1, Wv1, Ws1, Wq2, Wk2, Wv2, Ws2};
  for (int i = 0; i < 8; ++i)
    prep_w_kernel<<<blkW128, 256, 0, stream>>>(Wsrc[i], wtN[i], 128);
  prep_w_kernel<<<blkW64, 256, 0, stream>>>(We1, wtE1, 64);
  prep_w_kernel<<<blkW64, 256, 0, stream>>>(We2, wtE2, 64);

  // ---- Layer 1 ----
  wmma_gemm_kernel<128, 4, false><<<gemmN, 256, 0, stream>>>(
      xbf, N, wtN[0], wtN[1], wtN[2], wtN[3], bq1, bk1, bv1, bs1, qb, kb, vb, hb);
  wmma_gemm_kernel<64, 1, true><<<gemmE, 256, 0, stream>>>(
      eabf, E, wtE1, nullptr, nullptr, nullptr, nullptr, nullptr, nullptr, nullptr,
      ek, nullptr, nullptr, nullptr);
  init_nodes_kernel<<<blkN, 256, 0, stream>>>(amax, deno, N);
  edge_alpha_kernel<<<blkE8, 256, 0, stream>>>(srcI, dstI, qb, kb, ek, alph, amax, E, scale);
  edge_softmax_kernel<<<blkE, 256, 0, stream>>>(dstI, alph, amax, deno, E);
  edge_agg_kernel<<<blkE8, 256, 0, stream>>>(srcI, dstI, vb, ek, alph, deno, hb, E);
  // h = relu(h) packed to bf16 for the layer-2 GEMMs
  cvt_bf16_kernel<<<blkXq, 256, 0, stream>>>(hb, hbf, xq, 1);

  // ---- Layer 2 ----
  wmma_gemm_kernel<128, 4, false><<<gemmN, 256, 0, stream>>>(
      hbf, N, wtN[4], wtN[5], wtN[6], wtN[7], bq2, bk2, bv2, bs2, qb, kb, vb, outp);
  wmma_gemm_kernel<64, 1, true><<<gemmE, 256, 0, stream>>>(
      eabf, E, wtE2, nullptr, nullptr, nullptr, nullptr, nullptr, nullptr, nullptr,
      ek, nullptr, nullptr, nullptr);
  init_nodes_kernel<<<blkN, 256, 0, stream>>>(amax, deno, N);
  edge_alpha_kernel<<<blkE8, 256, 0, stream>>>(srcI, dstI, qb, kb, ek, alph, amax, E, scale);
  edge_softmax_kernel<<<blkE, 256, 0, stream>>>(dstI, alph, amax, deno, E);
  edge_agg_kernel<<<blkE8, 256, 0, stream>>>(srcI, dstI, vb, ek, alph, deno, outp, E);
}
